// SwinTransformerBlock3D_80728205295835
// MI455X (gfx1250) — matrix operand order repaired and checked by reading, not compiled, not run on hardware
//
#include <hip/hip_runtime.h>

// ---------------------------------------------------------------------------
// Swin-3D block for MI455X (gfx1250): wave32 + V_WMMA_F32_16X16X32_F16
// All WMMA operands staged in fragment-friendly layouts (2x ds_load_b128 per
// fragment); attention bias+mask precomputed into an L2-resident table.
// ---------------------------------------------------------------------------

typedef __attribute__((ext_vector_type(16))) _Float16 v16h;
typedef __attribute__((ext_vector_type(8)))  _Float16 v8h;
typedef __attribute__((ext_vector_type(4)))  _Float16 v4h;
typedef __attribute__((ext_vector_type(8)))  float    v8f;

#define WMMA(a, b, c) __builtin_amdgcn_wmma_f32_16x16x32_f16( \
    false, (a), false, (b), (short)0, (c), false, false)

// Problem constants
#define DD      4
#define HH      56
#define WW_     56
#define CC      128
#define NTOK    98            // tokens per window (2*7*7)
#define NPAD    112           // 7 * 16
#define NWIN    1024          // B * 128 windows
#define TOK     100352        // B*D*H*W = NWIN*NTOK
#define SCALE_Q 0.17677669529663687f   // 32^-0.5

// Workspace layout (bytes). Weights stored TRANSPOSED (N-major, f16).
static constexpr size_t OFF_WQ = 0;
static constexpr size_t OFF_WP = OFF_WQ + (size_t)384 * 128 * 2;
static constexpr size_t OFF_W1 = OFF_WP + (size_t)128 * 128 * 2;
static constexpr size_t OFF_W2 = OFF_W1 + (size_t)512 * 128 * 2;
static constexpr size_t OFF_XW = OFF_W2 + (size_t)128 * 512 * 2;
static constexpr size_t OFF_Q  = OFF_XW + (size_t)TOK * 128 * 2;
static constexpr size_t OFF_K  = OFF_Q  + (size_t)TOK * 128 * 2;
static constexpr size_t OFF_V  = OFF_K  + (size_t)TOK * 128 * 2;
static constexpr size_t OFF_AO = OFF_V  + (size_t)TOK * 128 * 2;
static constexpr size_t OFF_X1 = OFF_AO + (size_t)TOK * 128 * 2;
static constexpr size_t OFF_H2 = OFF_X1 + (size_t)TOK * 128 * 4;
static constexpr size_t OFF_BM = OFF_H2 + (size_t)TOK * 128 * 2;  // 8*4*112*112 f32

// ---------------------------------------------------------------------------
// WMMA fragment loaders (ISA 7.12.2 layouts, wave32) — fully vectorized.
// ---------------------------------------------------------------------------
__device__ __forceinline__ v16h load_frag_a(const _Float16* s, int ld) {
  const int lane = threadIdx.x & 31;
  const _Float16* r = s + (lane & 15) * ld + 8 * (lane >> 4);
  const v8h lo = *reinterpret_cast<const v8h*>(r);
  const v8h hi = *reinterpret_cast<const v8h*>(r + 16);
  return __builtin_shufflevector(lo, hi, 0, 1, 2, 3, 4, 5, 6, 7,
                                 8, 9, 10, 11, 12, 13, 14, 15);
}

__device__ __forceinline__ v16h load_frag_bt(const _Float16* s, int ld) {
  const int lane = threadIdx.x & 31;
  const _Float16* r = s + (lane & 15) * ld + 8 * (lane >> 4);
  const v8h lo = *reinterpret_cast<const v8h*>(r);
  const v8h hi = *reinterpret_cast<const v8h*>(r + 16);
  return __builtin_shufflevector(lo, hi, 0, 1, 2, 3, 4, 5, 6, 7,
                                 8, 9, 10, 11, 12, 13, 14, 15);
}

// shifted-window data-gather coords (cyclic roll by -shift)
__device__ __forceinline__ int src_d(int zi, int td) { return (zi * 2 + td + 1) & 3; }
__device__ __forceinline__ int src_h(int yi, int th) { return (yi * 7 + th + 3) % 56; }
__device__ __forceinline__ int src_w(int xi, int tw) { return (xi * 7 + tw + 3) % 56; }

// ---------------------------------------------------------------------------
// K0a: fp32 -> f16 weight conversion + transpose to N-major
// ---------------------------------------------------------------------------
__global__ void k_cvt_weights(const float* __restrict__ qkv_w,
                              const float* __restrict__ proj_w,
                              const float* __restrict__ fc1_w,
                              const float* __restrict__ fc2_w,
                              _Float16* __restrict__ wq, _Float16* __restrict__ wp,
                              _Float16* __restrict__ w1, _Float16* __restrict__ w2) {
  const int i = blockIdx.x * blockDim.x + threadIdx.x;
  if (i < 384 * 128) {                       // qkv_w (128 x 384) -> wq[n*128+k]
    const int n = i >> 7, k = i & 127;
    wq[i] = (_Float16)qkv_w[k * 384 + n];
  }
  if (i < 128 * 128) {                       // proj_w (128 x 128) -> wp[n*128+k]
    const int n = i >> 7, k = i & 127;
    wp[i] = (_Float16)proj_w[k * 128 + n];
  }
  if (i < 512 * 128) {                       // fc1_w (128 x 512) -> w1[n*128+k]
    const int n = i >> 7, k = i & 127;
    w1[i] = (_Float16)fc1_w[k * 512 + n];
  }
  if (i < 128 * 512) {                       // fc2_w (512 x 128) -> w2[n*512+k]
    const int n = i >> 9, k = i & 511;
    w2[i] = (_Float16)fc2_w[k * 128 + n];
  }
}

// ---------------------------------------------------------------------------
// K0b: combined rel-pos-bias + shift-mask table: bm[pat][head][m][n]
//   pat bits: (zi==1)<<2 | (yi==7)<<1 | (xi==7).  Regions are defined on the
//   SHIFTED-image coords (d' = zi*2+td etc.), so per-dim equality reduces to:
//   d: md!=nd (iff wrapping), h/w: (mh<4)!=(nh<4).
//   Pad entries (m or n >= 98) hold -1e30 -> folds softmax masking in.
// ---------------------------------------------------------------------------
__global__ void k_build_bm(const float* __restrict__ rpb, float* __restrict__ bm) {
  const int idx = blockIdx.x * blockDim.x + threadIdx.x;   // 8*4*112*112 exact
  const int n = idx % NPAD;
  const int m = (idx / NPAD) % NPAD;
  const int head = (idx / (NPAD * NPAD)) & 3;
  const int pat = idx / (4 * NPAD * NPAD);
  float v;
  if (m < NTOK && n < NTOK) {
    const int md = m / 49, mh = (m / 7) % 7, mw = m % 7;
    const int nd = n / 49, nh = (n / 7) % 7, nw = n % 7;
    const int bidx = (md - nd + 1) * 169 + (mh - nh + 6) * 13 + (mw - nw + 6);
    v = rpb[bidx * 4 + head];
    bool same = true;
    if ((pat & 4) && (md != nd)) same = false;
    if ((pat & 2) && ((mh < 4) != (nh < 4))) same = false;
    if ((pat & 1) && ((mw < 4) != (nw < 4))) same = false;
    if (!same) v -= 100.0f;
  } else {
    v = -1e30f;
  }
  bm[idx] = v;
}

// ---------------------------------------------------------------------------
// K1: LayerNorm1 + cyclic shift + window partition -> xw f16 (win, tok, C)
// ---------------------------------------------------------------------------
__global__ __launch_bounds__(256) void k_ln1_window(
    const float* __restrict__ x, const float* __restrict__ g,
    const float* __restrict__ b, _Float16* __restrict__ xw) {
  const int lane = threadIdx.x & 31;
  const int gi = blockIdx.x * 8 + (threadIdx.x >> 5);   // window-token index
  const int win = gi / NTOK, t = gi % NTOK;
  const int bb = win >> 7, wis = win & 127;
  const int zi = wis >> 6, yi = (wis >> 3) & 7, xi = wis & 7;
  const int td = t / 49, th = (t / 7) % 7, tw = t % 7;
  const size_t row = (((size_t)(bb * DD + src_d(zi, td)) * HH + src_h(yi, th)) *
                      WW_ + src_w(xi, tw)) * CC;
  const float4 v = reinterpret_cast<const float4*>(x + row)[lane];
  float s1 = v.x + v.y + v.z + v.w;
  float s2 = v.x * v.x + v.y * v.y + v.z * v.z + v.w * v.w;
#pragma unroll
  for (int off = 16; off; off >>= 1) {
    s1 += __shfl_xor(s1, off, 32);
    s2 += __shfl_xor(s2, off, 32);
  }
  const float mu = s1 * (1.0f / 128.0f);
  const float rs = rsqrtf(s2 * (1.0f / 128.0f) - mu * mu + 1e-5f);
  const int c = lane * 4;
  v4h o;
  o[0] = (_Float16)((v.x - mu) * rs * g[c + 0] + b[c + 0]);
  o[1] = (_Float16)((v.y - mu) * rs * g[c + 1] + b[c + 1]);
  o[2] = (_Float16)((v.z - mu) * rs * g[c + 2] + b[c + 2]);
  o[3] = (_Float16)((v.w - mu) * rs * g[c + 3] + b[c + 3]);
  *reinterpret_cast<v4h*>(xw + (size_t)gi * CC + c) = o;
}

// ---------------------------------------------------------------------------
// K2: QKV GEMM  (TOK x 128) @ (128 x 384) + bias -> Q/K (row-major heads),
//     V channel-major.  Block tile 64x128, 8 waves as 4(M) x 2(N).
// ---------------------------------------------------------------------------
__global__ __launch_bounds__(256) void k_gemm_qkv(
    const _Float16* __restrict__ xw, const _Float16* __restrict__ wq,
    const float* __restrict__ qkv_b, _Float16* __restrict__ qh,
    _Float16* __restrict__ kh, _Float16* __restrict__ vh) {
  __shared__ _Float16 As[64 * 128];     // (m, k) row-major
  __shared__ _Float16 Bs[128 * 128];    // (n, k) N-major
  const int tid = threadIdx.x;
  const int m0 = blockIdx.x * 64;
  const int n0 = blockIdx.y * 128;
  {
    const uint4* srcA = reinterpret_cast<const uint4*>(xw + (size_t)m0 * 128);
    const uint4* srcB = reinterpret_cast<const uint4*>(wq + (size_t)n0 * 128);
    uint4* dA = reinterpret_cast<uint4*>(As);
    uint4* dB = reinterpret_cast<uint4*>(Bs);
    for (int i = tid; i < 64 * 128 / 8; i += 256) dA[i] = srcA[i];
    for (int i = tid; i < 128 * 128 / 8; i += 256) dB[i] = srcB[i];
  }
  __syncthreads();
  const int wid = tid >> 5, lane = tid & 31, ln = lane & 15, hh = lane >> 4;
  const int wm = wid >> 1, wn = wid & 1;
  v8f acc[4] = {};
#pragma unroll
  for (int kc = 0; kc < 4; ++kc) {
    const v16h a = load_frag_a(As + (wm * 16) * 128 + kc * 32, 128);
#pragma unroll
    for (int nt = 0; nt < 4; ++nt) {
      const v16h bf = load_frag_bt(Bs + (wn * 64 + nt * 16) * 128 + kc * 32, 128);
      acc[nt] = WMMA(a, bf, acc[nt]);
    }
  }
#pragma unroll
  for (int nt = 0; nt < 4; ++nt) {
#pragma unroll
    for (int r = 0; r < 8; ++r) {
      const int gm = m0 + wm * 16 + r + 8 * hh;
      const int gn = n0 + wn * 64 + nt * 16 + ln;
      const float val = acc[nt][r] + qkv_b[gn];
      const int sec = gn >> 7, cin = gn & 127;
      const int head = cin >> 5, ch = cin & 31;
      const int win = gm / NTOK, row = gm % NTOK;
      const size_t wh = (size_t)(win * 4 + head);
      if (sec == 0)      qh[(wh * NTOK + row) * 32 + ch] = (_Float16)(val * SCALE_Q);
      else if (sec == 1) kh[(wh * NTOK + row) * 32 + ch] = (_Float16)val;
      else               vh[(wh * 32 + ch) * NTOK + row] = (_Float16)val; // ch-major
    }
  }
}

// ---------------------------------------------------------------------------
// K3: attention per (window, head). 7 waves, each owning 16 rows.
// ---------------------------------------------------------------------------
__global__ __launch_bounds__(224) void k_attn(
    const _Float16* __restrict__ qh, const _Float16* __restrict__ kh,
    const _Float16* __restrict__ vh, const float* __restrict__ bm,
    _Float16* __restrict__ ao) {
  __shared__ _Float16 Qs[NPAD * 32];    // (tok, ch)  — A operand
  __shared__ _Float16 Ks[NPAD * 32];    // (tok, ch)  — B^T operand (N-major)
  __shared__ _Float16 Vs[32 * 128];     // (ch, tok)  — B operand N-major, padded
  __shared__ _Float16 Ps[NPAD * 128];   // probs — A operand
  const int win = blockIdx.x >> 2, head = blockIdx.x & 3;
  const int tid = threadIdx.x;
  const size_t baseqk = ((size_t)(win * 4 + head)) * NTOK * 32;
  const size_t basev  = ((size_t)(win * 4 + head)) * 32 * NTOK;
  for (int i = tid; i < NPAD * 32; i += 224) {
    const int r = i >> 5;
    Qs[i] = (r < NTOK) ? qh[baseqk + i] : (_Float16)0;
    Ks[i] = (r < NTOK) ? kh[baseqk + i] : (_Float16)0;
  }
  for (int i = tid; i < 32 * 128; i += 224) {
    const int ch = i >> 7, tok = i & 127;
    Vs[i] = (tok < NTOK) ? vh[basev + ch * NTOK + tok] : (_Float16)0;
  }
  __syncthreads();

  const int wid = tid >> 5, lane = tid & 31, ln = lane & 15, hh = lane >> 4;
  const int m0 = wid * 16;
  const int wis = win & 127;
  const int zi = wis >> 6, yi = (wis >> 3) & 7, xi = wis & 7;
  const int pat = ((zi == 1) << 2) | ((yi == 7) << 1) | (xi == 7);
  const float* bmrow = bm + ((size_t)(pat * 4 + head)) * NPAD * NPAD;

  v8f acc[7] = {};
  const v16h aq = load_frag_a(Qs + m0 * 32, 32);
#pragma unroll
  for (int nt = 0; nt < 7; ++nt) {
    const v16h bk = load_frag_bt(Ks + (nt * 16) * 32, 32);
    acc[nt] = WMMA(aq, bk, acc[nt]);
  }

  // bias + mask (+ -1e30 padding) from the precomputed table: one load/elem
#pragma unroll
  for (int nt = 0; nt < 7; ++nt) {
    const int n = nt * 16 + ln;
#pragma unroll
    for (int r = 0; r < 8; ++r) {
      const int m = m0 + r + 8 * hh;
      acc[nt][r] += bmrow[m * NPAD + n];
    }
  }

  // row-wise softmax over the 16-lane half holding each row
#pragma unroll
  for (int r = 0; r < 8; ++r) {
    float mx = -1e30f;
#pragma unroll
    for (int nt = 0; nt < 7; ++nt) mx = fmaxf(mx, acc[nt][r]);
#pragma unroll
    for (int off = 1; off < 16; off <<= 1) mx = fmaxf(mx, __shfl_xor(mx, off, 32));
    float sum = 0.0f;
#pragma unroll
    for (int nt = 0; nt < 7; ++nt) {
      const float p = __expf(acc[nt][r] - mx);
      acc[nt][r] = p;
      sum += p;
    }
#pragma unroll
    for (int off = 1; off < 16; off <<= 1) sum += __shfl_xor(sum, off, 32);
    const float inv = 1.0f / sum;
    const int m = m0 + r + 8 * hh;
#pragma unroll
    for (int nt = 0; nt < 7; ++nt)
      Ps[m * 128 + nt * 16 + ln] = (_Float16)(acc[nt][r] * inv);
    Ps[m * 128 + 112 + ln] = (_Float16)0;   // zero K-pad 112..127
  }
  __syncthreads();

  // out = P (16x128) @ V (128x32); V is N-major (ch, tok)
  v8f o0 = {}, o1 = {};
#pragma unroll
  for (int kc = 0; kc < 4; ++kc) {
    const v16h ap = load_frag_a(Ps + m0 * 128 + kc * 32, 128);
    const v16h b0 = load_frag_bt(Vs + kc * 32, 128);
    const v16h b1 = load_frag_bt(Vs + 16 * 128 + kc * 32, 128);
    o0 = WMMA(ap, b0, o0);
    o1 = WMMA(ap, b1, o1);
  }
#pragma unroll
  for (int r = 0; r < 8; ++r) {
    const int m = m0 + r + 8 * hh;
    if (m < NTOK) {
      const size_t off = ((size_t)(win * NTOK + m)) * 128 + head * 32;
      ao[off + ln]      = (_Float16)o0[r];
      ao[off + 16 + ln] = (_Float16)o1[r];
    }
  }
}

// ---------------------------------------------------------------------------
// K4: proj GEMM + bias + window-reverse/un-shift + residual -> x1 fp32
// ---------------------------------------------------------------------------
__global__ __launch_bounds__(256) void k_proj_residual(
    const _Float16* __restrict__ ao, const _Float16* __restrict__ wp,
    const float* __restrict__ proj_b, const float* __restrict__ x,
    float* __restrict__ x1) {
  __shared__ _Float16 As[128 * 128];
  __shared__ _Float16 Bs[128 * 128];    // N-major
  const int tid = threadIdx.x;
  const int m0 = blockIdx.x * 128;
  for (int i = tid; i < 128 * 128 / 8; i += 256) {
    reinterpret_cast<uint4*>(As)[i] =
        reinterpret_cast<const uint4*>(ao + (size_t)m0 * 128)[i];
    reinterpret_cast<uint4*>(Bs)[i] = reinterpret_cast<const uint4*>(wp)[i];
  }
  __syncthreads();
  const int wid = tid >> 5, lane = tid & 31, ln = lane & 15, hh = lane >> 4;
  v8f acc[8] = {};
#pragma unroll
  for (int kc = 0; kc < 4; ++kc) {
    const v16h a = load_frag_a(As + (wid * 16) * 128 + kc * 32, 128);
#pragma unroll
    for (int nt = 0; nt < 8; ++nt) {
      const v16h bf = load_frag_bt(Bs + (nt * 16) * 128 + kc * 32, 128);
      acc[nt] = WMMA(a, bf, acc[nt]);
    }
  }
#pragma unroll
  for (int r = 0; r < 8; ++r) {
    const int gm = m0 + wid * 16 + r + 8 * hh;
    const int win = gm / NTOK, t = gm % NTOK;
    const int bb = win >> 7, wis = win & 127;
    const int zi = wis >> 6, yi = (wis >> 3) & 7, xi = wis & 7;
    const int td = t / 49, th = (t / 7) % 7, tw = t % 7;
    const size_t row =
        (((size_t)(bb * DD + src_d(zi, td)) * HH + src_h(yi, th)) * WW_ +
         src_w(xi, tw)) * CC;
#pragma unroll
    for (int nt = 0; nt < 8; ++nt) {
      const int col = nt * 16 + ln;
      x1[row + col] = acc[nt][r] + proj_b[col] + x[row + col];
    }
  }
}

// ---------------------------------------------------------------------------
// K5: LayerNorm2 -> h2 f16 (token-major)
// ---------------------------------------------------------------------------
__global__ __launch_bounds__(256) void k_ln2(
    const float* __restrict__ x1, const float* __restrict__ g,
    const float* __restrict__ b, _Float16* __restrict__ h2) {
  const int lane = threadIdx.x & 31;
  const int gi = blockIdx.x * 8 + (threadIdx.x >> 5);
  const float4 v = reinterpret_cast<const float4*>(x1 + (size_t)gi * CC)[lane];
  float s1 = v.x + v.y + v.z + v.w;
  float s2 = v.x * v.x + v.y * v.y + v.z * v.z + v.w * v.w;
#pragma unroll
  for (int off = 16; off; off >>= 1) {
    s1 += __shfl_xor(s1, off, 32);
    s2 += __shfl_xor(s2, off, 32);
  }
  const float mu = s1 * (1.0f / 128.0f);
  const float rs = rsqrtf(s2 * (1.0f / 128.0f) - mu * mu + 1e-5f);
  const int c = lane * 4;
  v4h o;
  o[0] = (_Float16)((v.x - mu) * rs * g[c + 0] + b[c + 0]);
  o[1] = (_Float16)((v.y - mu) * rs * g[c + 1] + b[c + 1]);
  o[2] = (_Float16)((v.z - mu) * rs * g[c + 2] + b[c + 2]);
  o[3] = (_Float16)((v.w - mu) * rs * g[c + 3] + b[c + 3]);
  *reinterpret_cast<v4h*>(h2 + (size_t)gi * CC + c) = o;
}

// ---------------------------------------------------------------------------
// K6: fused MLP: fc1 + exact GELU (hidden resident in LDS) + fc2 + residual
// ---------------------------------------------------------------------------
__global__ __launch_bounds__(256) void k_mlp(
    const _Float16* __restrict__ h2, const _Float16* __restrict__ w1,
    const float* __restrict__ b1, const _Float16* __restrict__ w2,
    const float* __restrict__ b2, const float* __restrict__ x1,
    float* __restrict__ out) {
  __shared__ _Float16 As[64 * 128];    // 16 KB (m, k)
  __shared__ _Float16 Ws[128 * 128];   // 32 KB (n, k) N-major, reused fc1/fc2
  __shared__ _Float16 Hs[64 * 512];    // 64 KB hidden, LDS-resident
  const int tid = threadIdx.x;
  const int m0 = blockIdx.x * 64;
  for (int i = tid; i < 64 * 128 / 8; i += 256)
    reinterpret_cast<uint4*>(As)[i] =
        reinterpret_cast<const uint4*>(h2 + (size_t)m0 * 128)[i];
  const int wid = tid >> 5, lane = tid & 31, ln = lane & 15, hh = lane >> 4;
  const int wm = wid >> 1, wn = wid & 1;

  // ---- fc1 + GELU, 4 chunks of 128 hidden cols ----
  for (int nc = 0; nc < 4; ++nc) {
    __syncthreads();
    for (int i = tid; i < 128 * 128 / 8; i += 256)
      reinterpret_cast<uint4*>(Ws)[i] =
          reinterpret_cast<const uint4*>(w1 + (size_t)(nc * 128) * 128)[i];
    __syncthreads();
    v8f acc[4] = {};
#pragma unroll
    for (int kc = 0; kc < 4; ++kc) {
      const v16h a = load_frag_a(As + (wm * 16) * 128 + kc * 32, 128);
#pragma unroll
      for (int nt = 0; nt < 4; ++nt) {
        const v16h bf = load_frag_bt(Ws + (wn * 64 + nt * 16) * 128 + kc * 32, 128);
        acc[nt] = WMMA(a, bf, acc[nt]);
      }
    }
#pragma unroll
    for (int nt = 0; nt < 4; ++nt) {
#pragma unroll
      for (int r = 0; r < 8; ++r) {
        const int ml = wm * 16 + r + 8 * hh;
        const int col = nc * 128 + wn * 64 + nt * 16 + ln;
        float v = acc[nt][r] + b1[col];
        v = 0.5f * v * (1.0f + erff(v * 0.70710678118654752f));  // exact GELU
        Hs[ml * 512 + col] = (_Float16)v;
      }
    }
  }

  // ---- fc2, K = 512 in 4 outer chunks; w2 is (n, k) with ldk = 512 ----
  v8f acc2[4] = {};
  for (int ko = 0; ko < 4; ++ko) {
    __syncthreads();
    for (int i = tid; i < 128 * 128 / 8; i += 256) {
      const int n = i >> 4, c = i & 15;
      reinterpret_cast<uint4*>(Ws)[i] =
          reinterpret_cast<const uint4*>(w2 + (size_t)n * 512 + ko * 128)[c];
    }
    __syncthreads();
#pragma unroll
    for (int kc = 0; kc < 4; ++kc) {
      const v16h a = load_frag_a(Hs + (wm * 16) * 512 + ko * 128 + kc * 32, 512);
#pragma unroll
      for (int nt = 0; nt < 4; ++nt) {
        const v16h bf = load_frag_bt(Ws + (wn * 64 + nt * 16) * 128 + kc * 32, 128);
        acc2[nt] = WMMA(a, bf, acc2[nt]);
      }
    }
  }
#pragma unroll
  for (int nt = 0; nt < 4; ++nt) {
#pragma unroll
    for (int r = 0; r < 8; ++r) {
      const int gm = m0 + wm * 16 + r + 8 * hh;
      const int col = wn * 64 + nt * 16 + ln;
      out[(size_t)gm * 128 + col] = acc2[nt][r] + b2[col] + x1[(size_t)gm * 128 + col];
    }
  }
}

// ---------------------------------------------------------------------------
extern "C" void kernel_launch(void* const* d_in, const int* in_sizes, int n_in,
                              void* d_out, int out_size, void* d_ws, size_t ws_size,
                              hipStream_t stream) {
  (void)in_sizes; (void)n_in; (void)out_size; (void)ws_size;
  const float* x      = (const float*)d_in[0];
  const float* n1g    = (const float*)d_in[1];
  const float* n1b    = (const float*)d_in[2];
  const float* qkv_w  = (const float*)d_in[3];
  const float* qkv_b  = (const float*)d_in[4];
  const float* rpb    = (const float*)d_in[5];
  const float* proj_w = (const float*)d_in[6];
  const float* proj_b = (const float*)d_in[7];
  const float* n2g    = (const float*)d_in[8];
  const float* n2b    = (const float*)d_in[9];
  const float* fc1_w  = (const float*)d_in[10];
  const float* fc1_b  = (const float*)d_in[11];
  const float* fc2_w  = (const float*)d_in[12];
  const float* fc2_b  = (const float*)d_in[13];
  float* out = (float*)d_out;

  char* ws = (char*)d_ws;
  _Float16* wq  = (_Float16*)(ws + OFF_WQ);
  _Float16* wp  = (_Float16*)(ws + OFF_WP);
  _Float16* w1  = (_Float16*)(ws + OFF_W1);
  _Float16* w2  = (_Float16*)(ws + OFF_W2);
  _Float16* xw  = (_Float16*)(ws + OFF_XW);
  _Float16* qhp = (_Float16*)(ws + OFF_Q);
  _Float16* khp = (_Float16*)(ws + OFF_K);
  _Float16* vhp = (_Float16*)(ws + OFF_V);
  _Float16* ao  = (_Float16*)(ws + OFF_AO);
  float*    x1  = (float*)   (ws + OFF_X1);
  _Float16* h2  = (_Float16*)(ws + OFF_H2);
  float*    bm  = (float*)   (ws + OFF_BM);

  k_cvt_weights<<<256, 256, 0, stream>>>(qkv_w, proj_w, fc1_w, fc2_w, wq, wp, w1, w2);
  k_build_bm<<<(8 * 4 * NPAD * NPAD) / 256, 256, 0, stream>>>(rpb, bm);
  k_ln1_window<<<TOK / 8, 256, 0, stream>>>(x, n1g, n1b, xw);
  k_gemm_qkv<<<dim3(TOK / 64, 3), 256, 0, stream>>>(xw, wq, qkv_b, qhp, khp, vhp);
  k_attn<<<NWIN * 4, 224, 0, stream>>>(qhp, khp, vhp, bm, ao);
  k_proj_residual<<<TOK / 128, 256, 0, stream>>>(ao, wp, proj_b, x, x1);
  k_ln2<<<TOK / 8, 256, 0, stream>>>(x1, n2g, n2b, h2);
  k_mlp<<<TOK / 64, 256, 0, stream>>>(h2, w1, fc1_b, w2, fc2_b, x1, out);
}
